// CausalSelfAttention_24034636988909
// MI455X (gfx1250) — matrix operand order repaired and checked by reading, not compiled, not run on hardware
//
#include <hip/hip_runtime.h>
#include <math.h>

// Problem constants (from reference)
#define B_  4
#define S_  1024
#define DM  1024
#define NH  16
#define HD  64
#define NR  64
#define RK  8
#define SR  32
#define NB  16

typedef __attribute__((ext_vector_type(16))) _Float16 v16h;
typedef __attribute__((ext_vector_type(8)))  _Float16 v8h;
typedef __attribute__((ext_vector_type(8)))  float    v8f;

__device__ __forceinline__ v8f wmma16(v16h a, v16h b, v8f c) {
  // D = A(16x32 f16) * B(32x16 f16) + C(16x16 f32)
  return __builtin_amdgcn_wmma_f32_16x16x32_f16(false, a, false, b, (short)0, c,
                                                false, false);
}

__device__ __forceinline__ v16h cat8(v8h lo, v8h hi) {
  return __builtin_shufflevector(lo, hi, 0, 1, 2, 3, 4, 5, 6, 7,
                                         8, 9, 10, 11, 12, 13, 14, 15);
}

// ---------------------------------------------------------------------------
// Rule-aware projection: out = x@si@so + gate * blockdiag-adapter(x; rule)
// One workgroup (256 threads) per token.
//   mode 0: fp32 out [B,S,1024]            (o-projection -> d_out)
//   mode 1: f16  out [B,H,S,64]            (q, k; rope applied if apply_rope)
//   mode 2: f16  out [B,H,64,S] transposed (v; enables contiguous B-frags)
// ---------------------------------------------------------------------------
__global__ __launch_bounds__(256) void rule_proj_kernel(
    const float* __restrict__ xin, const int* __restrict__ rules,
    const float* __restrict__ si, const float* __restrict__ so,
    const float* __restrict__ ri, const float* __restrict__ ro,
    const float* __restrict__ lg,
    float* __restrict__ out32, _Float16* __restrict__ out16,
    int apply_rope, int mode)
{
  __shared__ float xs[DM];
  __shared__ float ph[8 * SR];
  __shared__ float hs[SR];
  __shared__ float hb[NB * RK];
  __shared__ float sel[NB];
  __shared__ float outv[DM];

  const int tok  = blockIdx.x;
  const int b    = tok / S_;
  const int s    = tok % S_;
  const int tid  = threadIdx.x;
  const int rule = rules[tok];

  for (int d = tid; d < DM; d += 256) xs[d] = xin[(size_t)tok * DM + d];
  __syncthreads();

  // shared-rank partials: hs[r] = sum_d x[d]*si[d][r]
  {
    const int r = tid & 31, c = tid >> 5;   // c in 0..7, 128-wide chunks
    float sum = 0.f;
    const int d0 = c * 128;
    for (int d = 0; d < 128; ++d) sum += xs[d0 + d] * si[(size_t)(d0 + d) * SR + r];
    ph[c * SR + r] = sum;
  }
  // adapter hidden: hb[bb][rr] = sum_{s2<64} x[bb*64+s2] * ri[rule][s2][rr]
  if (tid < NB * RK) {
    const int bb = tid >> 3, rr = tid & 7;
    float sum = 0.f;
    const float* rip = ri + (size_t)rule * HD * RK;
    for (int s2 = 0; s2 < HD; ++s2) sum += xs[bb * HD + s2] * rip[s2 * RK + rr];
    hb[tid] = sum;
  }
  __syncthreads();
  if (tid < SR) {
    float sum = 0.f;
    for (int c = 0; c < 8; ++c) sum += ph[c * SR + tid];
    hs[tid] = sum;
  }
  if (tid == 0) {
    // sel = softmax(lg[rule]*sqrt(16))
    float e[NB], m = -1e38f;
    for (int i = 0; i < NB; ++i) { e[i] = lg[rule * NB + i] * 4.0f; m = fmaxf(m, e[i]); }
    float ssum = 0.f;
    for (int i = 0; i < NB; ++i) { e[i] = __expf(e[i] - m); ssum += e[i]; }
    for (int i = 0; i < NB; ++i) sel[i] = e[i] / ssum;
  }
  __syncthreads();

  for (int d = tid; d < DM; d += 256) {
    float o = 0.f;
    for (int r = 0; r < SR; ++r) o += hs[r] * so[(size_t)r * DM + d];
    const int bb = d >> 6, dd = d & 63;
    float a = 0.f;
    const float* rop = ro + (size_t)rule * RK * HD;
    for (int rr = 0; rr < RK; ++rr) a += hb[bb * RK + rr] * rop[rr * HD + dd];
    outv[d] = o + a * sel[bb];
  }
  __syncthreads();

  for (int d = tid; d < DM; d += 256) {
    const int hh = d >> 6, dd = d & 63;
    float val = outv[d];
    if (apply_rope) {
      const int i = dd >> 1;
      const float freq = __expf(-(float)(2 * i) * (9.210340371976184f / 64.0f)); // ln(1e4)/64
      const float ang = (float)s * freq;
      float sn, cs;
      __sincosf(ang, &sn, &cs);
      if ((dd & 1) == 0) val = outv[d] * cs - outv[d + 1] * sn;
      else               val = outv[d] * cs + outv[d - 1] * sn;
    }
    if (mode == 0)
      out32[(size_t)tok * DM + d] = val;                               // [B,S,1024] fp32
    else if (mode == 1)
      out16[(((size_t)b * NH + hh) * S_ + s) * HD + dd] = (_Float16)val;  // [B,H,S,64]
    else
      out16[(((size_t)b * NH + hh) * HD + dd) * S_ + s] = (_Float16)val;  // [B,H,64,S]
  }
}

// ---------------------------------------------------------------------------
// Rule routing mask. Affinity of row (b,q) only takes 64 values (one per key
// rule). Build per-row rule histogram, exact top-32 threshold via cumulative
// counts (matches "masked >= thr" tie semantics), pack bits: word w covers
// keys [32w, 32w+31].
// ---------------------------------------------------------------------------
__global__ __launch_bounds__(64) void mask_kernel(
    const int* __restrict__ rules,
    const float* __restrict__ router_q, const float* __restrict__ router_k,
    unsigned* __restrict__ maskw)
{
  __shared__ float Arow[NR];
  __shared__ int   cnt[NR];
  __shared__ float cand[NR];
  __shared__ float thr_s;

  const int bq = blockIdx.x;
  const int b = bq / S_, q = bq % S_;
  const int tid = threadIdx.x;
  const int rq = rules[(size_t)b * S_ + q];

  if (tid < NR) {
    float qv[RK], kv[RK], nq = 0.f, nk = 0.f, dot = 0.f;
    for (int i = 0; i < RK; ++i) { qv[i] = router_q[rq * RK + i];  nq += qv[i] * qv[i]; }
    for (int i = 0; i < RK; ++i) { kv[i] = router_k[tid * RK + i]; nk += kv[i] * kv[i]; }
    nq = fmaxf(sqrtf(nq), 1e-12f);
    nk = fmaxf(sqrtf(nk), 1e-12f);
    for (int i = 0; i < RK; ++i) dot += qv[i] * kv[i];
    Arow[tid] = dot / (nq * nk) * 2.0794415416798357f;  // /tau == *ln(8)
    cnt[tid] = 0;
  }
  __syncthreads();
  for (int k = tid; k <= q; k += 64)
    atomicAdd(&cnt[rules[(size_t)b * S_ + k]], 1);
  __syncthreads();
  if (tid < NR) {
    const float a = Arow[tid];
    int cum = 0;
    for (int i = 0; i < NR; ++i) if (Arow[i] >= a) cum += cnt[i];
    cand[tid] = (cum >= 32) ? a : -1e38f;
  }
  __syncthreads();
  if (tid == 0) {
    float t = -1e38f;
    for (int i = 0; i < NR; ++i) t = fmaxf(t, cand[i]);
    if (q + 1 < 32) t = -1e38f;  // fewer than top_k causal entries -> keep all
    thr_s = t;
  }
  __syncthreads();
  const float thr = thr_s;
  if (tid < 32) {
    unsigned bits = 0;
    const int kb = tid * 32;
    for (int j = 0; j < 32; ++j) {
      const int k = kb + j;
      if (k <= q && Arow[rules[(size_t)b * S_ + k]] >= thr) bits |= (1u << j);
    }
    maskw[(size_t)bq * 32 + tid] = bits;
  }
}

// ---------------------------------------------------------------------------
// Flash attention: one wave per 16-query tile, stream 32-key tiles.
// Fragment packing (A and B share the contiguous k order):
//   A (16x32 f16): lane half=L>>4, row m=L&15; slot j -> K=16*half+j
//                  => one contiguous 16-half run (two b128 loads)
//   B (32x16 f16): lane col n=L&15; slot j -> K=16*half+j
//                  => one contiguous 16-half run (two b128 loads)
//   C (16x16 f32): vgpr i -> row m=i+8*half; col n=L&15
// K is stored [pos][dim] (B-frag run = dims of one key row),
// V is stored transposed [dim][pos] (B-frag run = positions of one dim row).
// ---------------------------------------------------------------------------
__global__ __launch_bounds__(32) void attn_kernel(
    const _Float16* __restrict__ qw, const _Float16* __restrict__ kw,
    const _Float16* __restrict__ vtw, const unsigned* __restrict__ maskw,
    float* __restrict__ aout)
{
  __shared__ unsigned mw[16];
  __shared__ _Float16 Pl[16 * 32] __attribute__((aligned(16)));

  const int blk = blockIdx.x;
  const int qt  = blk % (S_ / 16);
  const int bh  = blk / (S_ / 16);
  const int h   = bh % NH, b = bh / NH;
  const int q0  = qt * 16;

  const int lane = threadIdx.x;
  const int half = lane >> 4;
  const int ln   = lane & 15;

  const _Float16* qb  = qw  + (size_t)bh * S_ * HD;   // [S][64]
  const _Float16* kb  = kw  + (size_t)bh * S_ * HD;   // [S][64]
  const _Float16* vtb = vtw + (size_t)bh * HD * S_;   // [64][S]

  // Q A-fragments: two head-dim K-chunks of 32, each = two contiguous v8h
  v16h Aq[2];
  #pragma unroll
  for (int c = 0; c < 2; ++c) {
    const _Float16* qr = qb + (size_t)(q0 + ln) * HD + 32 * c + 16 * half;
    Aq[c] = cat8(*(const v8h*)qr, *(const v8h*)(qr + 8));
  }

  float rmax[8], rsum[8];
  v8f   O[4];
  const v8f zf = {};
  #pragma unroll
  for (int i = 0; i < 8; ++i) { rmax[i] = -INFINITY; rsum[i] = 0.f; }
  #pragma unroll
  for (int c = 0; c < 4; ++c) O[c] = zf;

  const int ntiles = (q0 + 15) / 32 + 1;
  for (int t = 0; t < ntiles; ++t) {
    const int kbase = t * 32;

    // prefetch next tile of K rows and V-transposed rows (global_prefetch_b8)
    if (t + 1 < ntiles) {
      const int nb = kbase + 32;
      __builtin_prefetch(kb + (size_t)(nb + lane) * HD, 0, 0);
      __builtin_prefetch(vtb + (size_t)lane * S_ + nb, 0, 0);
      __builtin_prefetch(vtb + (size_t)(lane + 32) * S_ + nb, 0, 0);
    }

    __syncthreads();
    if (lane < 16)
      mw[lane] = maskw[(size_t)(b * S_ + q0 + lane) * 32 + (kbase >> 5)];
    __syncthreads();

    // scores for the two 16-wide key groups (each: 2 WMMAs over head dim)
    v8f Sg[2];
    #pragma unroll
    for (int g = 0; g < 2; ++g) {
      v8f acc = zf;
      #pragma unroll
      for (int c = 0; c < 2; ++c) {
        const _Float16* kr =
            kb + (size_t)(kbase + 16 * g + ln) * HD + 32 * c + 16 * half;
        const v16h Bk = cat8(*(const v8h*)kr, *(const v8h*)(kr + 8));
        acc = wmma16(Aq[c], Bk, acc);
      }
      Sg[g] = acc;
    }

    // online softmax update (row = i + 8*half, reduced over 16-lane halves)
    #pragma unroll
    for (int i = 0; i < 8; ++i) {
      const int m = i + 8 * half;
      const int qq = q0 + m;
      float s0, s1;
      {
        const bool ok = ((kbase + ln) <= qq) && ((mw[m] >> ln) & 1u);
        s0 = ok ? Sg[0][i] * 0.125f : -INFINITY;
      }
      {
        const bool ok = ((kbase + 16 + ln) <= qq) && ((mw[m] >> (16 + ln)) & 1u);
        s1 = ok ? Sg[1][i] * 0.125f : -INFINITY;
      }
      float v = fmaxf(s0, s1);
      #pragma unroll
      for (int off = 1; off < 16; off <<= 1) v = fmaxf(v, __shfl_xor(v, off, 32));
      const float nm = fmaxf(rmax[i], v);
      const float a  = (rmax[i] > -1e37f) ? __expf(rmax[i] - nm) : 0.f;
      const float p0 = (s0 > -1e37f) ? __expf(s0 - nm) : 0.f;
      const float p1 = (s1 > -1e37f) ? __expf(s1 - nm) : 0.f;
      Pl[m * 32 + ln]      = (_Float16)p0;
      Pl[m * 32 + 16 + ln] = (_Float16)p1;
      float ps = p0 + p1;
      #pragma unroll
      for (int off = 1; off < 16; off <<= 1) ps += __shfl_xor(ps, off, 32);
      rsum[i] = rsum[i] * a + ps;
      rmax[i] = nm;
      #pragma unroll
      for (int c = 0; c < 4; ++c) O[c][i] *= a;
    }
    __syncthreads();

    // P (C layout in LDS) -> A-fragment: two contiguous 8-half LDS reads
    v16h Pa;
    {
      const _Float16* pr = Pl + ln * 32;
      Pa = cat8(*(const v8h*)(pr + 16 * half),
                *(const v8h*)(pr + 16 * half + 8));
    }

    // O += P(16x32) * V(32x64): four 16-wide head-dim chunks, from V^T rows
    #pragma unroll
    for (int c = 0; c < 4; ++c) {
      const _Float16* vr = vtb + (size_t)(16 * c + ln) * S_ + kbase + 16 * half;
      const v16h Bv = cat8(*(const v8h*)vr, *(const v8h*)(vr + 8));
      O[c] = wmma16(Pa, Bv, O[c]);
    }
  }

  // normalize and write [B,S,H*64] (matches transpose(0,2,1,3).reshape)
  #pragma unroll
  for (int c = 0; c < 4; ++c) {
    #pragma unroll
    for (int i = 0; i < 8; ++i) {
      const int m = i + 8 * half;
      const float val = O[c][i] / rsum[i];
      aout[((size_t)b * S_ + q0 + m) * DM + h * HD + 16 * c + ln] = val;
    }
  }
}

// ---------------------------------------------------------------------------
// Host-side orchestration. Workspace layout:
//   qws/kws: f16 [B,H,S,64]   (4,194,304 halves = 8 MB each)
//   vtws:    f16 [B,H,64,S]   (8 MB)
//   aws:     f32 [B,S,1024]   (16 MB)
//   maskws:  u32 [B,S,32]     (512 KB)
// Total ~40.5 MB (fully L2-resident on MI455X's 192 MB L2).
// ---------------------------------------------------------------------------
extern "C" void kernel_launch(void* const* d_in, const int* in_sizes, int n_in,
                              void* d_out, int out_size, void* d_ws, size_t ws_size,
                              hipStream_t stream)
{
  (void)in_sizes; (void)n_in; (void)out_size; (void)ws_size;

  const float* x     = (const float*)d_in[0];
  const int*   rules = (const int*)d_in[1];
  const float* qsi = (const float*)d_in[2],  *qso = (const float*)d_in[3];
  const float* qri = (const float*)d_in[4],  *qro = (const float*)d_in[5];
  const float* qlg = (const float*)d_in[6];
  const float* ksi = (const float*)d_in[7],  *kso = (const float*)d_in[8];
  const float* kri = (const float*)d_in[9],  *kro = (const float*)d_in[10];
  const float* klg = (const float*)d_in[11];
  const float* vsi = (const float*)d_in[12], *vso = (const float*)d_in[13];
  const float* vri = (const float*)d_in[14], *vro = (const float*)d_in[15];
  const float* vlg = (const float*)d_in[16];
  const float* osi = (const float*)d_in[17], *oso = (const float*)d_in[18];
  const float* ori = (const float*)d_in[19], *oro = (const float*)d_in[20];
  const float* olg = (const float*)d_in[21];
  const float* router_q = (const float*)d_in[22];
  const float* router_k = (const float*)d_in[23];

  const size_t QKV = (size_t)B_ * NH * S_ * HD;   // 4,194,304 elements
  _Float16* qws  = (_Float16*)d_ws;
  _Float16* kws  = qws + QKV;
  _Float16* vtws = kws + QKV;
  float*    aws  = (float*)(vtws + QKV);
  unsigned* maskws = (unsigned*)(aws + (size_t)B_ * S_ * DM);

  const dim3 grdT(B_ * S_), blkT(256);
  rule_proj_kernel<<<grdT, blkT, 0, stream>>>(x, rules, qsi, qso, qri, qro, qlg,
                                              nullptr, qws, 1, 1);
  rule_proj_kernel<<<grdT, blkT, 0, stream>>>(x, rules, ksi, kso, kri, kro, klg,
                                              nullptr, kws, 1, 1);
  rule_proj_kernel<<<grdT, blkT, 0, stream>>>(x, rules, vsi, vso, vri, vro, vlg,
                                              nullptr, vtws, 0, 2);

  mask_kernel<<<dim3(B_ * S_), dim3(64), 0, stream>>>(rules, router_q, router_k, maskws);

  attn_kernel<<<dim3(B_ * NH * (S_ / 16)), dim3(32), 0, stream>>>(qws, kws, vtws,
                                                                  maskws, aws);

  rule_proj_kernel<<<grdT, blkT, 0, stream>>>(aws, rules, osi, oso, ori, oro, olg,
                                              (float*)d_out, nullptr, 0, 0);
}
